// GCN_47579647705319
// MI455X (gfx1250) — matrix operand order repaired and checked by reading, not compiled, not run on hardware
//
#include <hip/hip_runtime.h>
#include <hip/hip_fp16.h>

// =====================================================================
// 4-layer GCN for MI455X (gfx1250, wave32, WMMA).
//
// Roofline: GEMMs ~7.6 GFLOP total -> trivial on v_wmma f16->f32; the
// dominant cost is the edge gather/scatter (~2-3 GB, entirely L2-resident
// given 192 MB L2). Activations are kept in f16 (BN re-normalizes every
// layer so quantization is benign); weights are pre-swizzled into WMMA
// fragment layout so the GEMM inner loop is 1 vector LDS read + 1 WMMA
// per K-step (no scalar ds_load_u16 assembly).
//
// Workspace (~135 MB): bufA/bufB N*256 f32, h16 N*256 f16, norm E f32,
// deg/dinv N, wpad 256*256 f16 (swizzled), BN scratch.
// =====================================================================

typedef __attribute__((ext_vector_type(16))) _Float16 v16h;
typedef __attribute__((ext_vector_type(8)))  _Float16 v8h;
typedef __attribute__((ext_vector_type(8)))  float    v8f;

__device__ __forceinline__ void atomAddF(float* p, float v) {
  unsafeAtomicAdd(p, v);   // native global_atomic_add_f32
}

// ---------------- degree / symmetric norm ----------------
__global__ void deg_count_kernel(const int* __restrict__ dst,
                                 unsigned* __restrict__ deg, int E) {
  int e = blockIdx.x * blockDim.x + threadIdx.x;
  if (e < E) atomicAdd(&deg[dst[e]], 1u);
}

__global__ void dinv_kernel(const unsigned* __restrict__ deg,
                            float* __restrict__ dinv, int N) {
  int i = blockIdx.x * blockDim.x + threadIdx.x;
  if (i < N) dinv[i] = rsqrtf((float)(deg[i] + 1u));   // +1 = self loop
}

__global__ void norm_kernel(const int* __restrict__ src, const int* __restrict__ dst,
                            const float* __restrict__ dinv,
                            float* __restrict__ nrm, int E) {
  int e = blockIdx.x * blockDim.x + threadIdx.x;
  if (e < E) nrm[e] = dinv[src[e]] * dinv[dst[e]];
}

// ---------------- activation f32 -> f16 ----------------
__global__ void f32_to_f16_kernel(const float* __restrict__ X,
                                  _Float16* __restrict__ Y, size_t total) {
  size_t i = (size_t)blockIdx.x * blockDim.x + threadIdx.x;
  if (i < total) Y[i] = (_Float16)X[i];
}

// ---- weight f32 -> f16, swizzled into WMMA B-fragment layout ----
// Fragment (nt,kt) occupies 32 lanes x 16 halves contiguously:
//   Wp[((nt*KT + kt)*32 + lane)*16 + e] = W[k, col],
//   k = kt*32 + (lane>>4)*16 + e, col = nt*16 + (lane&15)   (contiguous k order, same as A)
__global__ void convert_w_swz_kernel(const float* __restrict__ W,
                                     _Float16* __restrict__ Wp,
                                     int K, int F, int Np, int KT) {
  int o = blockIdx.x * blockDim.x + threadIdx.x;
  if (o >= K * Np) return;
  int e    = o & 15;
  int lane = (o >> 4) & 31;
  int fi   = o >> 9;
  int kt   = fi % KT;
  int nt   = fi / KT;
  int k    = kt * 32 + (lane >> 4) * 16 + e;
  int col  = nt * 16 + (lane & 15);
  Wp[o] = (col < F) ? (_Float16)W[(size_t)k * F + col] : (_Float16)0.f;
}

// ---------------- WMMA GEMM: C[N,Np] = A[N,K](f16) * W(f16, swizzled) -
// One 16-row tile per wave; A fragments in registers for all K-steps
// (2x global_load_b128 each); B fragments are single 32B vector LDS
// reads from the pre-swizzled weight image. Wave-uniform control flow
// keeps EXEC all-ones as WMMA requires.
#define GEMM_WAVES 4

template <int K, int Np>
__global__ void __launch_bounds__(32 * GEMM_WAVES)
gemm_wmma_kernel(const _Float16* __restrict__ A, const _Float16* __restrict__ Bswz,
                 float* __restrict__ C, int nTiles) {
  extern __shared__ _Float16 Bs[];
  constexpr int KT = K / 32;
  constexpr int NT = Np / 16;

  // stage swizzled weights (vector copies)
  {
    const v8h* s8 = (const v8h*)Bswz;
    v8h* d8 = (v8h*)Bs;
    for (int i = threadIdx.x; i < (K * Np) / 8; i += blockDim.x) d8[i] = s8[i];
  }
  __syncthreads();

  const int wid  = threadIdx.x >> 5;
  const int lane = threadIdx.x & 31;
  const int m    = lane & 15;
  const int hi   = lane >> 4;
  const v16h* Bfrag = (const v16h*)Bs;   // [NT*KT][32 lanes] of 32B fragments

  for (int t = blockIdx.x * GEMM_WAVES + wid; t < nTiles;
       t += gridDim.x * GEMM_WAVES) {
    // A fragments (contiguous k order, same as B): lane m, two 8-half runs per K-step
    v16h afrag[KT];
    const v8h* arow8 = (const v8h*)(A + (size_t)(t * 16 + m) * K);
#pragma unroll
    for (int kt = 0; kt < KT; ++kt) {
      v8h lo = arow8[kt * 4 + 2 * hi];       // K = kt*32 + hi*16 + 0..7
      v8h hh = arow8[kt * 4 + 2 * hi + 1];   // K = kt*32 + hi*16 + 8..15
      afrag[kt] = __builtin_shufflevector(lo, hh, 0, 1, 2, 3, 4, 5, 6, 7,
                                          8, 9, 10, 11, 12, 13, 14, 15);
    }
    for (int nt = 0; nt < NT; ++nt) {
      v8f c = {};
#pragma unroll
      for (int kt = 0; kt < KT; ++kt) {
        v16h b = Bfrag[(nt * KT + kt) * 32 + lane];
        c = __builtin_amdgcn_wmma_f32_16x16x32_f16(
            false, afrag[kt], false, b, (short)0, c, false, false);
      }
#pragma unroll
      for (int r = 0; r < 8; ++r) {      // C: VGPR r -> row r + 8*hi
        C[(size_t)(t * 16 + hi * 8 + r) * Np + nt * 16 + m] = c[r];
      }
    }
  }
}

// ---------------- acc = bias + dinv^2 * xw   (self-loop + bias) -------
__global__ void init_acc_kernel(const float* __restrict__ xw,
                                const float* __restrict__ dinv,
                                const float* __restrict__ bias,
                                int Freal, int F,
                                float* __restrict__ acc, int N) {
  size_t i = (size_t)blockIdx.x * blockDim.x + threadIdx.x;
  size_t total = (size_t)N * F;
  if (i >= total) return;
  int f = (int)(i % F);
  int r = (int)(i / F);
  float dv = dinv[r];
  acc[i] = ((f < Freal) ? bias[f] : 0.f) + dv * dv * xw[i];
}

// ---------------- edge scatter: acc[d] += norm[e] * xw[s] -------------
__global__ void scatter_kernel(const float* __restrict__ xw,
                               const int* __restrict__ src,
                               const int* __restrict__ dst,
                               const float* __restrict__ nrm,
                               float* __restrict__ acc, int F, int E) {
  int e = blockIdx.x * blockDim.y + threadIdx.y;
  if (e >= E) return;
  int f = threadIdx.x;
  int s = src[e], d = dst[e];
  float w = nrm[e];
  atomAddF(&acc[(size_t)d * F + f], xw[(size_t)s * F + f] * w);
}

// ---------------- BatchNorm stats (thread-per-feature, coalesced) -----
template <int F>
__global__ void __launch_bounds__(256)
bn_stats_kernel(const float* __restrict__ X, int N, float* __restrict__ sums) {
  const int f    = threadIdx.x % F;      // F divides 256
  const int ro   = threadIdx.x / F;
  const int step = 256 / F;
  float s = 0.f, ss = 0.f;
  for (int r = blockIdx.x * step + ro; r < N; r += gridDim.x * step) {
    float v = X[(size_t)r * F + f];
    s += v; ss += v * v;
  }
  atomAddF(&sums[f], s);
  atomAddF(&sums[F + f], ss);
}

__global__ void bn_finalize_kernel(const float* __restrict__ sums,
                                   const float* __restrict__ g,
                                   const float* __restrict__ be,
                                   float invN, int F,
                                   float* __restrict__ scale,
                                   float* __restrict__ shift) {
  int f = blockIdx.x * blockDim.x + threadIdx.x;
  if (f >= F) return;
  float mean = sums[f] * invN;
  float var  = sums[F + f] * invN - mean * mean;   // biased var (jnp.var)
  float sc   = g[f] * rsqrtf(var + 1e-5f);
  scale[f] = sc;
  shift[f] = be[f] - mean * sc;
}

// ------- fused BN apply + PReLU, emitting f16 for the next GEMM -------
__global__ void bn_prelu_f16_kernel(const float* __restrict__ X,
                                    _Float16* __restrict__ H,
                                    const float* __restrict__ scale,
                                    const float* __restrict__ shift,
                                    const float* __restrict__ aP,
                                    int fmask, size_t total) {
  size_t i = (size_t)blockIdx.x * blockDim.x + threadIdx.x;
  if (i >= total) return;
  int f = (int)i & fmask;                 // F is a power of two
  float a = aP[0];
  float y = X[i] * scale[f] + shift[f];
  H[i] = (_Float16)((y >= 0.f) ? y : a * y);
}

// ---------------- row-wise log_softmax over first C of F cols ---------
__global__ void logsoftmax_kernel(const float* __restrict__ acc,
                                  float* __restrict__ out,
                                  int N, int F, int C) {
  int r = blockIdx.x * blockDim.x + threadIdx.x;
  if (r >= N) return;
  const float* x = acc + (size_t)r * F;
  float mx = x[0];
  for (int j = 1; j < C; ++j) mx = fmaxf(mx, x[j]);
  float se = 0.f;
  for (int j = 0; j < C; ++j) se += expf(x[j] - mx);
  float lse = logf(se) + mx;
  for (int j = 0; j < C; ++j) out[(size_t)r * C + j] = x[j] - lse;
}

// =====================================================================
extern "C" void kernel_launch(void* const* d_in, const int* in_sizes, int n_in,
                              void* d_out, int out_size, void* d_ws, size_t ws_size,
                              hipStream_t stream) {
  const float* x   = (const float*)d_in[0];
  const int*   ei  = (const int*)d_in[1];
  const float* W1  = (const float*)d_in[2];
  const float* b1  = (const float*)d_in[3];
  const float* g1  = (const float*)d_in[4];
  const float* be1 = (const float*)d_in[5];
  const float* W2  = (const float*)d_in[6];
  const float* b2  = (const float*)d_in[7];
  const float* g2  = (const float*)d_in[8];
  const float* be2 = (const float*)d_in[9];
  const float* W3  = (const float*)d_in[10];
  const float* b3  = (const float*)d_in[11];
  const float* g3  = (const float*)d_in[12];
  const float* be3 = (const float*)d_in[13];
  const float* W4  = (const float*)d_in[14];
  const float* b4  = (const float*)d_in[15];
  const float* aP  = (const float*)d_in[16];

  const int N  = in_sizes[0] / 128;   // 50000 (multiple of 16)
  const int E  = in_sizes[1] / 2;     // 800000
  const int nT = N / 16;
  const int* srcI = ei;               // edge_index[0] : gather source
  const int* dstI = ei + E;           // edge_index[1] : scatter dest

  char* w = (char*)d_ws;
  auto alloc = [&](size_t bytes) -> char* {
    char* p = w; w += (bytes + 255) & ~(size_t)255; return p;
  };
  float*    bufA = (float*)alloc((size_t)N * 256 * 4);   // GEMM out / xw
  float*    bufB = (float*)alloc((size_t)N * 256 * 4);   // accumulator
  _Float16* h16  = (_Float16*)alloc((size_t)N * 256 * 2);// f16 activations
  float*    nrm  = (float*)alloc((size_t)E * 4);
  unsigned* deg  = (unsigned*)alloc((size_t)N * 4);
  float*    dinv = (float*)alloc((size_t)N * 4);
  _Float16* wpad = (_Float16*)alloc(256 * 256 * 2);      // swizzled weights
  float*    bns  = (float*)alloc(2 * 256 * 4);
  float*    bsc  = (float*)alloc(256 * 4);
  float*    bsh  = (float*)alloc(256 * 4);
  (void)ws_size; (void)n_in; (void)out_size;

  const float invN = 1.0f / (float)N;
  const int GG = 512;   // GEMM blocks (4 waves each)

  // ---- shared normalization (once per launch) ----
  hipMemsetAsync(deg, 0, (size_t)N * 4, stream);
  deg_count_kernel<<<(E + 255) / 256, 256, 0, stream>>>(dstI, deg, E);
  dinv_kernel<<<(N + 255) / 256, 256, 0, stream>>>(deg, dinv, N);
  norm_kernel<<<(E + 255) / 256, 256, 0, stream>>>(srcI, dstI, dinv, nrm, E);

  // ================= Layer 1: 128 -> 256 =================
  f32_to_f16_kernel<<<(int)(((size_t)N * 128 + 255) / 256), 256, 0, stream>>>(
      x, h16, (size_t)N * 128);
  convert_w_swz_kernel<<<(128 * 256 + 255) / 256, 256, 0, stream>>>(
      W1, wpad, 128, 256, 256, 128 / 32);
  gemm_wmma_kernel<128, 256><<<GG, 128, 128 * 256 * 2, stream>>>(h16, wpad, bufA, nT);
  init_acc_kernel<<<(int)(((size_t)N * 256 + 255) / 256), 256, 0, stream>>>(
      bufA, dinv, b1, 256, 256, bufB, N);
  { dim3 blk(256, 1); scatter_kernel<<<E, blk, 0, stream>>>(bufA, srcI, dstI, nrm, bufB, 256, E); }
  hipMemsetAsync(bns, 0, 2 * 256 * 4, stream);
  bn_stats_kernel<256><<<256, 256, 0, stream>>>(bufB, N, bns);
  bn_finalize_kernel<<<1, 256, 0, stream>>>(bns, g1, be1, invN, 256, bsc, bsh);
  bn_prelu_f16_kernel<<<(int)(((size_t)N * 256 + 255) / 256), 256, 0, stream>>>(
      bufB, h16, bsc, bsh, aP, 255, (size_t)N * 256);

  // ================= Layer 2: 256 -> 128 =================
  convert_w_swz_kernel<<<(256 * 128 + 255) / 256, 256, 0, stream>>>(
      W2, wpad, 256, 128, 128, 256 / 32);
  gemm_wmma_kernel<256, 128><<<GG, 128, 256 * 128 * 2, stream>>>(h16, wpad, bufA, nT);
  init_acc_kernel<<<(int)(((size_t)N * 128 + 255) / 256), 256, 0, stream>>>(
      bufA, dinv, b2, 128, 128, bufB, N);
  { dim3 blk(128, 2); scatter_kernel<<<(E + 1) / 2, blk, 0, stream>>>(bufA, srcI, dstI, nrm, bufB, 128, E); }
  hipMemsetAsync(bns, 0, 2 * 256 * 4, stream);
  bn_stats_kernel<128><<<256, 256, 0, stream>>>(bufB, N, bns);
  bn_finalize_kernel<<<1, 256, 0, stream>>>(bns, g2, be2, invN, 128, bsc, bsh);
  bn_prelu_f16_kernel<<<(int)(((size_t)N * 128 + 255) / 256), 256, 0, stream>>>(
      bufB, h16, bsc, bsh, aP, 127, (size_t)N * 128);

  // ================= Layer 3: 128 -> 64 =================
  convert_w_swz_kernel<<<(128 * 64 + 255) / 256, 256, 0, stream>>>(
      W3, wpad, 128, 64, 64, 128 / 32);
  gemm_wmma_kernel<128, 64><<<GG, 128, 128 * 64 * 2, stream>>>(h16, wpad, bufA, nT);
  init_acc_kernel<<<(int)(((size_t)N * 64 + 255) / 256), 256, 0, stream>>>(
      bufA, dinv, b3, 64, 64, bufB, N);
  { dim3 blk(64, 4); scatter_kernel<<<(E + 3) / 4, blk, 0, stream>>>(bufA, srcI, dstI, nrm, bufB, 64, E); }
  hipMemsetAsync(bns, 0, 2 * 256 * 4, stream);
  bn_stats_kernel<64><<<256, 256, 0, stream>>>(bufB, N, bns);
  bn_finalize_kernel<<<1, 256, 0, stream>>>(bns, g3, be3, invN, 64, bsc, bsh);
  bn_prelu_f16_kernel<<<(int)(((size_t)N * 64 + 255) / 256), 256, 0, stream>>>(
      bufB, h16, bsc, bsh, aP, 63, (size_t)N * 64);

  // ================= Layer 4: 64 -> 40 (padded to 48) + log_softmax ===
  convert_w_swz_kernel<<<(64 * 48 + 255) / 256, 256, 0, stream>>>(
      W4, wpad, 64, 40, 48, 64 / 32);
  gemm_wmma_kernel<64, 48><<<GG, 128, 64 * 48 * 2, stream>>>(h16, wpad, bufA, nT);
  init_acc_kernel<<<(int)(((size_t)N * 48 + 255) / 256), 256, 0, stream>>>(
      bufA, dinv, b4, 40, 48, bufB, N);
  { dim3 blk(48, 5); scatter_kernel<<<(E + 4) / 5, blk, 0, stream>>>(bufA, srcI, dstI, nrm, bufB, 48, E); }
  logsoftmax_kernel<<<(N + 255) / 256, 256, 0, stream>>>(bufB, (float*)d_out, N, 48, 40);
}
